// BaselineModelSplitter_54503134986368
// MI455X (gfx1250) — matrix operand order repaired and checked by reading, not compiled, not run on hardware
//
#include <hip/hip_runtime.h>

typedef __attribute__((ext_vector_type(16))) _Float16 v16h;
typedef __attribute__((ext_vector_type(8)))  float    v8f;
typedef int v4i __attribute__((vector_size(16)));

#define BB 512
#define NN 64

#define ASG __attribute__((address_space(1)))
#define ASL __attribute__((address_space(3)))

#if defined(__has_builtin)
#if __has_builtin(__builtin_amdgcn_global_load_async_to_lds_b128)
#define HAVE_ASYNC 1
#endif
#if __has_builtin(__builtin_amdgcn_s_wait_asynccnt)
#define HAVE_WAIT_ASYNC 1
#endif
#endif

// ---------------- conversion kernels ----------------
__global__ void k_f32_to_f16(const float* __restrict__ s, _Float16* __restrict__ d, int n) {
  int i = blockIdx.x * 256 + threadIdx.x;
  if (i < n) d[i] = (_Float16)s[i];
}

// W is (K,N) row-major f32; Wt is (N,K) row-major f16
__global__ void k_xpose(const float* __restrict__ W, _Float16* __restrict__ Wt, int K, int N) {
  int i = blockIdx.x * 256 + threadIdx.x;
  if (i < K * N) {
    int k = i / N, n = i - k * N;
    Wt[(size_t)n * K + k] = (_Float16)W[i];
  }
}

// ---------------- WMMA GEMM: C = act(A(MxK) @ W(KxN) + bias [+ rowBias]) ----------------
// A: f16 row-major, lda halves. Wt: f16 (N x K) row-major, ldw halves.
// Grid: (N/64, M/64), block 128 threads (4 waves); wave w owns rows [16w,16w+16).
__global__ __launch_bounds__(128) void k_gemm(
    const _Float16* __restrict__ A, int lda,
    const _Float16* __restrict__ Wt, int ldw,
    const float* __restrict__ bias,
    const float* __restrict__ rowBias, int rbs,   // indexed [row>>6][col], may be null
    _Float16* __restrict__ C16, float* __restrict__ C32, int ldc,
    int K, int relu)
{
  __shared__ _Float16 lA[64 * 32] __attribute__((aligned(16)));
  __shared__ _Float16 lB[64 * 32] __attribute__((aligned(16)));

  const int bm   = blockIdx.y * 64;
  const int bn   = blockIdx.x * 64;
  const int tid  = threadIdx.x;
  const int lane = tid & 31;
  const int wave = tid >> 5;
  const int lr   = lane & 15;
  const int hi   = lane >> 4;
  const int r    = tid >> 1;          // 0..63 row of tile this thread loads
  const int part = (tid & 1) << 4;    // 0 or 16 halves

  v8f acc[4];
#pragma unroll
  for (int t = 0; t < 4; t++)
#pragma unroll
    for (int j = 0; j < 8; j++) acc[t][j] = 0.0f;

  const _Float16* gA = A  + (size_t)(bm + r) * lda + part;
  const _Float16* gB = Wt + (size_t)(bn + r) * ldw + part;
  _Float16* sA = &lA[r * 32 + part];
  _Float16* sB = &lB[r * 32 + part];

  for (int kk = 0; kk < K; kk += 32) {
    __syncthreads();
#ifdef HAVE_ASYNC
    // CDNA5 async tile staging: global -> LDS without touching VGPRs (ASYNCcnt)
    __builtin_amdgcn_global_load_async_to_lds_b128(
        (ASG v4i*)(gA + kk),     (ASL v4i*)sA,       0, 0);
    __builtin_amdgcn_global_load_async_to_lds_b128(
        (ASG v4i*)(gA + kk + 8), (ASL v4i*)(sA + 8), 0, 0);
    __builtin_amdgcn_global_load_async_to_lds_b128(
        (ASG v4i*)(gB + kk),     (ASL v4i*)sB,       0, 0);
    __builtin_amdgcn_global_load_async_to_lds_b128(
        (ASG v4i*)(gB + kk + 8), (ASL v4i*)(sB + 8), 0, 0);
#ifdef HAVE_WAIT_ASYNC
    __builtin_amdgcn_s_wait_asynccnt(0);
#else
    asm volatile("s_wait_asynccnt 0" ::: "memory");
#endif
#else
    uint4 va0 = *(const uint4*)(gA + kk);
    uint4 va1 = *(const uint4*)(gA + kk + 8);
    uint4 vb0 = *(const uint4*)(gB + kk);
    uint4 vb1 = *(const uint4*)(gB + kk + 8);
    *(uint4*)sA       = va0;
    *(uint4*)(sA + 8) = va1;
    *(uint4*)sB       = vb0;
    *(uint4*)(sB + 8) = vb1;
#endif
    __syncthreads();

    // A fragment 16x32 (ISA layout: lanes 0-15 -> K 0-7 & 16-23; lanes 16-31 -> K 8-15 & 24-31)
    v16h af;
    {
      const _Float16* pa = &lA[(wave * 16 + lr) * 32 + hi * 8];
      ((uint4*)&af)[0] = *(const uint4*)pa;
      ((uint4*)&af)[1] = *(const uint4*)(pa + 16);
    }
#pragma unroll
    for (int nt = 0; nt < 4; nt++) {
      // B fragment 32x16: lane = column, halves = K (lanes 0-15: K 0-7 & 16-23, lanes 16-31: K 8-15 & 24-31)
      v16h bf;
      const _Float16* pb = &lB[(nt * 16 + lr) * 32 + hi * 8];
      ((uint4*)&bf)[0] = *(const uint4*)pb;
      ((uint4*)&bf)[1] = *(const uint4*)(pb + 16);
      acc[nt] = __builtin_amdgcn_wmma_f32_16x16x32_f16(false, af, false, bf,
                                                       (short)0, acc[nt], false, false);
    }
  }

  // C/D layout: lane L, elem i -> row = 16*wave + 8*(L>=16) + i, col = lr
#pragma unroll
  for (int nt = 0; nt < 4; nt++) {
    int col = bn + nt * 16 + lr;
    float bv = bias ? bias[col] : 0.0f;
#pragma unroll
    for (int i = 0; i < 8; i++) {
      int row = bm + wave * 16 + hi * 8 + i;
      float v = acc[nt][i] + bv;
      if (rowBias) v += rowBias[(size_t)(row >> 6) * rbs + col];
      if (relu && v < 0.0f) v = 0.0f;
      if (C16) C16[(size_t)row * ldc + col] = (_Float16)v;
      if (C32) C32[(size_t)row * ldc + col] = v;
    }
  }
}

// ---------------- small kernels ----------------
// s[row] = X2[row,:] . w + b   (K = 128, N = 1)
__global__ void k_dot128(const _Float16* __restrict__ X, const float* __restrict__ w,
                         const float* __restrict__ b, float* __restrict__ s, int M) {
  int r = blockIdx.x * 256 + threadIdx.x;
  if (r < M) {
    const _Float16* x = X + (size_t)r * 128;
    float acc = b[0];
#pragma unroll 16
    for (int k = 0; k < 128; k++) acc += (float)x[k] * w[k];
    s[r] = acc;
  }
}

// masked 4-way softmax, product, renormalize, weighted bbox sums. One block of 64 per batch.
__global__ __launch_bounds__(64) void k_combine(
    const float* __restrict__ s4, const int* __restrict__ len,
    const float* __restrict__ bbox, float* __restrict__ sub, float* __restrict__ emb)
{
  int b = blockIdx.x;
  int n = threadIdx.x;
  __shared__ float sm[64];
  bool valid = n < len[b];
  float p[4];
#pragma unroll
  for (int o = 0; o < 4; o++) {
    float x = valid ? s4[(size_t)o * BB * NN + b * NN + n] : -1e30f;
    sm[n] = x; __syncthreads();
    for (int off = 32; off > 0; off >>= 1) { if (n < off) sm[n] = fmaxf(sm[n], sm[n + off]); __syncthreads(); }
    float mx = sm[0]; __syncthreads();
    float e = valid ? __expf(x - mx) : 0.0f;
    sm[n] = e; __syncthreads();
    for (int off = 32; off > 0; off >>= 1) { if (n < off) sm[n] += sm[n + off]; __syncthreads(); }
    float ssum = sm[0]; __syncthreads();
    p[o] = e / ssum;
  }
  float sr = p[0] * p[1], obr = p[2] * p[3];
  sm[n] = sr; __syncthreads();
  for (int off = 32; off > 0; off >>= 1) { if (n < off) sm[n] += sm[n + off]; __syncthreads(); }
  float subsum = sm[0]; __syncthreads();
  sm[n] = obr; __syncthreads();
  for (int off = 32; off > 0; off >>= 1) { if (n < off) sm[n] += sm[n + off]; __syncthreads(); }
  float objsum = sm[0]; __syncthreads();
  float subv = sr / subsum;
  float objv = obr / objsum;
  sub[b * NN + n] = subv;
  const float* bbp = bbox + ((size_t)b * NN + n) * 5;
  for (int k = 0; k < 5; k++) {
    sm[n] = objv * bbp[k]; __syncthreads();
    for (int off = 32; off > 0; off >>= 1) { if (n < off) sm[n] += sm[n + off]; __syncthreads(); }
    if (n == 0) emb[b * 13 + 3 + k] = sm[0];
    __syncthreads();
    sm[n] = subv * bbp[k]; __syncthreads();
    for (int off = 32; off > 0; off >>= 1) { if (n < off) sm[n] += sm[n + off]; __syncthreads(); }
    if (n == 0) emb[b * 13 + 8 + k] = sm[0];
    __syncthreads();
  }
}

// decoder final layer (128 -> 3) + softmax into emb[:, 0:3]
__global__ void k_dec3(const _Float16* __restrict__ D3, const float* __restrict__ W,
                       const float* __restrict__ bias, float* __restrict__ emb) {
  int b = blockIdx.x * 64 + threadIdx.x;
  if (b < BB) {
    float l[3];
#pragma unroll
    for (int j = 0; j < 3; j++) {
      float a = bias[j];
      for (int k = 0; k < 128; k++) a += (float)D3[(size_t)b * 128 + k] * W[k * 3 + j];
      l[j] = a;
    }
    float mx = fmaxf(l[0], fmaxf(l[1], l[2]));
    float e0 = __expf(l[0] - mx), e1 = __expf(l[1] - mx), e2 = __expf(l[2] - mx);
    float s = e0 + e1 + e2;
    emb[b * 13 + 0] = e0 / s;
    emb[b * 13 + 1] = e1 / s;
    emb[b * 13 + 2] = e2 / s;
  }
}

// pred layer 0 (K = 13)
__global__ void k_pred0(const float* __restrict__ emb, const float* __restrict__ W,
                        const float* __restrict__ bias, _Float16* __restrict__ P0) {
  int idx = blockIdx.x * 256 + threadIdx.x;
  if (idx < BB * 256) {
    int b = idx >> 8, j = idx & 255;
    float a = bias[j];
#pragma unroll
    for (int k = 0; k < 13; k++) a += emb[b * 13 + k] * W[k * 256 + j];
    P0[idx] = (_Float16)(a > 0.0f ? a : 0.0f);
  }
}

// pred final layer (256 -> 5) + tanh
__global__ void k_pred4(const _Float16* __restrict__ P3, const float* __restrict__ W,
                        const float* __restrict__ bias, float* __restrict__ predv) {
  int idx = blockIdx.x * 256 + threadIdx.x;
  if (idx < BB * 5) {
    int b = idx / 5, j = idx - b * 5;
    float a = bias[j];
    for (int k = 0; k < 256; k++) a += (float)P3[(size_t)b * 256 + k] * W[k * 5 + j];
    predv[idx] = tanhf(a);
  }
}

// final masked blend
__global__ void k_final(const float* __restrict__ sub, const float* __restrict__ bbox,
                        const float* __restrict__ predv, const int* __restrict__ len,
                        float* __restrict__ out) {
  int idx = blockIdx.x * 256 + threadIdx.x;
  if (idx < BB * NN * 5) {
    int k = idx % 5;
    int bn = idx / 5;
    int n = bn & (NN - 1);
    int b = bn >> 6;
    float s = sub[bn];
    float v = (1.0f - s) * bbox[idx] + s * predv[b * 5 + k];
    out[idx] = (n < len[b]) ? v : 0.0f;
  }
}

extern "C" void kernel_launch(void* const* d_in, const int* in_sizes, int n_in,
                              void* d_out, int out_size, void* d_ws, size_t ws_size,
                              hipStream_t stream)
{
  (void)in_sizes; (void)n_in; (void)out_size; (void)ws_size;
  const float* f_objects = (const float*)d_in[0];
  const float* f_instr   = (const float*)d_in[1];
  const float* bboxes    = (const float*)d_in[2];
  const int*   obj_len   = (const int*)d_in[3];
  auto P = [&](int i) -> const float* { return (const float*)d_in[i]; };

  char* basep = (char*)d_ws;
  size_t off = 0;
  auto alloc = [&](size_t bytes) -> void* {
    void* p = basep + off;
    off = (off + bytes + 255) & ~(size_t)255;
    return p;
  };

  _Float16* fobj16  = (_Float16*)alloc((size_t)BB * NN * 256 * 2);
  _Float16* instr16 = (_Float16*)alloc((size_t)BB * 512 * 2);
  _Float16* hbuf    = (_Float16*)alloc((size_t)BB * 512 * 2);
  _Float16* i256    = (_Float16*)alloc((size_t)BB * 256 * 2);
  float*    tbuf    = (float*)   alloc((size_t)BB * 256 * 4);
  _Float16* X0      = (_Float16*)alloc((size_t)BB * NN * 256 * 2);
  _Float16* X1      = (_Float16*)alloc((size_t)BB * NN * 256 * 2);
  _Float16* X2      = (_Float16*)alloc((size_t)BB * NN * 128 * 2);
  float*    s4      = (float*)   alloc((size_t)4 * BB * NN * 4);
  float*    subbuf  = (float*)   alloc((size_t)BB * NN * 4);
  float*    emb     = (float*)   alloc((size_t)BB * 13 * 4);
  _Float16* Pa      = (_Float16*)alloc((size_t)BB * 256 * 2);
  _Float16* Pb      = (_Float16*)alloc((size_t)BB * 512 * 2);
  _Float16* Pc      = (_Float16*)alloc((size_t)BB * 512 * 2);
  _Float16* Pd      = (_Float16*)alloc((size_t)BB * 256 * 2);
  float*    predv   = (float*)   alloc((size_t)BB * 5 * 4);
  _Float16* D0      = (_Float16*)alloc((size_t)BB * 512 * 2);
  _Float16* D1      = (_Float16*)alloc((size_t)BB * 256 * 2);
  _Float16* D2      = (_Float16*)alloc((size_t)BB * 256 * 2);
  _Float16* D3      = (_Float16*)alloc((size_t)BB * 128 * 2);
  // transposed f16 weight staging buffers (reused across sequential phases)
  _Float16* wt0     = (_Float16*)alloc((size_t)512 * 512 * 2);
  _Float16* wt1     = (_Float16*)alloc((size_t)512 * 256 * 2);
  _Float16* wt2     = (_Float16*)alloc((size_t)512 * 256 * 2);
  _Float16* wt3     = (_Float16*)alloc((size_t)256 * 256 * 2);
  _Float16* wt4     = (_Float16*)alloc((size_t)256 * 128 * 2);

  auto cvt = [&](const float* s, _Float16* d, int n) {
    k_f32_to_f16<<<(n + 255) / 256, 256, 0, stream>>>(s, d, n);
  };
  auto xp = [&](const float* W, _Float16* Wt, int K, int N) {
    k_xpose<<<(K * N + 255) / 256, 256, 0, stream>>>(W, Wt, K, N);
  };
  auto gemm = [&](const _Float16* A, int lda, const _Float16* Wt, int ldw,
                  const float* bias, const float* rowBias, int rbs,
                  _Float16* C16, float* C32, int ldc, int M, int K, int N, int relu) {
    dim3 g(N / 64, M / 64);
    k_gemm<<<g, 128, 0, stream>>>(A, lda, Wt, ldw, bias, rowBias, rbs, C16, C32, ldc, K, relu);
  };

  cvt(f_objects, fobj16, BB * NN * 256);
  cvt(f_instr, instr16, BB * 512);

  // params flat order: sub_map@4, attn_sub@8, sub_shape_map@16, attn_shape_sub@20,
  // obj_map@28, attn_obj@32, obj_shape_map@40, attn_shape_obj@44, decoder@52, pred@62
  const int mapIdx[4]  = {4, 16, 28, 40};
  const int attnIdx[4] = {8, 20, 32, 44};
  for (int i = 0; i < 4; i++) {
    int m = mapIdx[i], a = attnIdx[i];
    xp(P(m),     wt0, 512, 512);   // map W0
    xp(P(m + 2), wt1, 512, 256);   // map W1
    xp(P(a),     wt2, 512, 256);   // attn W0 (rows 0..255 = instr part, 256..511 = object part)
    xp(P(a + 2), wt3, 256, 256);   // attn W1
    xp(P(a + 4), wt4, 256, 128);   // attn W2
    // H = relu(instr @ Wm0 + bm0)
    gemm(instr16, 512, wt0, 512, P(m + 1), nullptr, 0, hbuf, nullptr, 512, BB, 512, 512, 1);
    // I = H @ Wm1 + bm1
    gemm(hbuf,    512, wt1, 512, P(m + 3), nullptr, 0, i256, nullptr, 256, BB, 512, 256, 0);
    // t = I @ Wa0_top + ba0   (per-batch row bias for layer 0)
    gemm(i256,    256, wt2, 512, P(a + 1), nullptr, 0, nullptr, tbuf, 256, BB, 256, 256, 0);
    // X0 = relu(f_objects @ Wa0_bot + t[b])
    gemm(fobj16,  256, wt2 + 256, 512, nullptr, tbuf, 256, X0, nullptr, 256, BB * NN, 256, 256, 1);
    // X1 = relu(X0 @ Wa1 + ba1)
    gemm(X0,      256, wt3, 256, P(a + 3), nullptr, 0, X1, nullptr, 256, BB * NN, 256, 256, 1);
    // X2 = relu(X1 @ Wa2 + ba2)
    gemm(X1,      256, wt4, 256, P(a + 5), nullptr, 0, X2, nullptr, 128, BB * NN, 256, 128, 1);
    // s = X2 @ wa3 + ba3
    k_dot128<<<(BB * NN + 255) / 256, 256, 0, stream>>>(X2, P(a + 6), P(a + 7),
                                                        s4 + (size_t)i * BB * NN, BB * NN);
  }

  // decoder MLP 512->512->256->256->128->3 (softmax)
  xp(P(52), wt0, 512, 512);
  xp(P(54), wt1, 512, 256);
  xp(P(56), wt3, 256, 256);
  xp(P(58), wt4, 256, 128);
  gemm(instr16, 512, wt0, 512, P(53), nullptr, 0, D0, nullptr, 512, BB, 512, 512, 1);
  gemm(D0, 512, wt1, 512, P(55), nullptr, 0, D1, nullptr, 256, BB, 512, 256, 1);
  gemm(D1, 256, wt3, 256, P(57), nullptr, 0, D2, nullptr, 256, BB, 256, 256, 1);
  gemm(D2, 256, wt4, 256, P(59), nullptr, 0, D3, nullptr, 128, BB, 256, 128, 1);
  k_dec3<<<BB / 64, 64, 0, stream>>>(D3, P(60), P(61), emb);

  // softmax products, renormalize, bbox-weighted locations -> emb[:,3:13], sub
  k_combine<<<BB, 64, 0, stream>>>(s4, obj_len, bboxes, subbuf, emb);

  // pred MLP 13->256->512->512->256->5 (tanh)
  k_pred0<<<(BB * 256 + 255) / 256, 256, 0, stream>>>(emb, P(62), P(63), Pa);
  xp(P(64), wt1, 256, 512);
  xp(P(66), wt0, 512, 512);
  xp(P(68), wt2, 512, 256);
  gemm(Pa, 256, wt1, 256, P(65), nullptr, 0, Pb, nullptr, 512, BB, 256, 512, 1);
  gemm(Pb, 512, wt0, 512, P(67), nullptr, 0, Pc, nullptr, 512, BB, 512, 512, 1);
  gemm(Pc, 512, wt2, 512, P(69), nullptr, 0, Pd, nullptr, 256, BB, 512, 256, 1);
  k_pred4<<<(BB * 5 + 255) / 256, 256, 0, stream>>>(Pd, P(70), P(71), predv);

  // final masked blend into output (B, N, 5) f32
  k_final<<<(BB * NN * 5 + 255) / 256, 256, 0, stream>>>(subbuf, bboxes, predv, obj_len,
                                                         (float*)d_out);
}
